// SparseMoELanguageModel_83803401880040
// MI455X (gfx1250) — matrix operand order repaired and checked by reading, not compiled, not run on hardware
//
#include <hip/hip_runtime.h>
#include <cstdint>
#include <cstddef>

// ---------------- model constants ----------------
#define BB   64
#define TT   48
#define MM   (BB*TT)        // 3072 tokens
#define DD   144
#define DP   160            // D padded to multiple of 32 for WMMA K
#define H4   576            // 4*D (already /32)
#define NHh  6
#define HDh  24
#define NEe  6
#define NLl  6
#define VV   50257

typedef __attribute__((ext_vector_type(16))) __bf16        v16bf;
typedef __attribute__((ext_vector_type(8)))  float         v8f;
typedef __attribute__((ext_vector_type(4)))  unsigned int  u32x4;

// ---------------- small device helpers ----------------
__device__ inline __bf16 f2bf(float f) {
    union { float f; uint32_t u; } in; in.f = f;
    uint32_t u = in.u;
    u += 0x7FFFu + ((u >> 16) & 1u);               // round-to-nearest-even
    union { unsigned short s; __bf16 b; } out;
    out.s = (unsigned short)(u >> 16);
    return out.b;
}

__device__ inline float geluf(float v) {
    return 0.5f * v * (1.0f + erff(v * 0.70710678118f));
}

__device__ inline float hashnorm(uint32_t s) {
    s ^= s >> 17; s *= 0xed5ad4bbu; s ^= s >> 11; s *= 0xac4c1b51u;
    s ^= s >> 15; s *= 0x31848babu; s ^= s >> 14;
    uint32_t s2 = s * 0x9E3779B9u + 0x7F4A7C15u;
    s2 ^= s2 >> 16; s2 *= 0x85ebca6bu; s2 ^= s2 >> 13;
    float u1 = ((s & 0x7FFFFFu) + 1u) * (1.0f / 8388608.0f);
    float u2 = (s2 & 0xFFFFFFu) * (1.0f / 16777216.0f);
    return sqrtf(-2.0f * logf(u1)) * cosf(6.28318530718f * u2);
}

// A-matrix fragment (16x32 bf16), row = lane&15, ISA 7.12.2 layout:
// lanes 0-15: halves 0-7 = K 0..7, halves 8-15 = K 16..23
// lanes16-31: halves 0-7 = K 8..15, halves 8-15 = K 24..31
__device__ inline v16bf load_fragA(const __bf16* p, int lane) {
    const int k0 = (lane >> 4) << 3;               // 0 or 8
    union { u32x4 u[2]; v16bf v; } f;
    f.u[0] = *(const u32x4*)(p + k0);
    f.u[1] = *(const u32x4*)(p + k0 + 16);
    return f.v;
}

// B-matrix fragment (32x16 bf16), col = lane&15; same K layout as A:
// lanes 0-15: K 0..7, 16..23; lanes 16-31: K 8..15, 24..31.
__device__ inline v16bf load_fragB(const __bf16* p, int lane) {
    const int k0 = (lane >> 4) << 3;               // 0 or 8
    union { u32x4 u[2]; v16bf v; } f;
    f.u[0] = *(const u32x4*)(p + k0);
    f.u[1] = *(const u32x4*)(p + k0 + 16);
    return f.v;
}

// ---------------- generic bf16 WMMA GEMM ----------------
// One wave computes a 16(M) x 64(N) strip: 1 A-frag reused by 4 WMMAs.
// CHUNKS = K/32 is a compile-time constant; the K loop fully unrolls into a
// single straight-line body (no loop-carried registers -> zero rotation moves;
// the scheduler hoists each chunk's load clauses above earlier WMMAs).
// C[m,n] = sum_k A[m,k]*W[n,k] + bias[n] (+resid) (gelu?) -> f32 and/or bf16
template<int CHUNKS>
__global__ __launch_bounds__(256)
void gemm_wmma_kernel(const __bf16* __restrict__ A, int lda,
                      const __bf16* __restrict__ W, int ldw,
                      const float* __restrict__ bias,
                      const float* __restrict__ resid, int ldr,
                      float* __restrict__ Cf, __bf16* __restrict__ Cb, int ldc,
                      int Mrows, int Nrows, int stripsN, int act) {
    const int wave = blockIdx.x * (blockDim.x >> 5) + (threadIdx.x >> 5);
    const int lane = threadIdx.x & 31;
    const int tM = (wave / stripsN) * 16;
    const int tN = (wave % stripsN) * 64;
    if (tM >= Mrows) return;                       // uniform per wave

    const int am = tM + (lane & 15);
    const __bf16* pa = A + (size_t)am * lda;
    const __bf16* pb[4];
    int bn[4];
    v8f acc[4];
#pragma unroll
    for (int j = 0; j < 4; ++j) {
        bn[j] = tN + 16 * j + (lane & 15);
        const int bnc = (bn[j] < Nrows) ? bn[j] : 0;
        pb[j] = W + (size_t)bnc * ldw;
        const float bv = bias ? bias[bnc] : 0.0f;
#pragma unroll
        for (int i = 0; i < 8; ++i) acc[j][i] = bv;
    }

#pragma unroll
    for (int c = 0; c < CHUNKS; ++c) {
        const int k = c << 5;
        v16bf a  = load_fragA(pa + k, lane);
        v16bf b0 = load_fragB(pb[0] + k, lane);
        v16bf b1 = load_fragB(pb[1] + k, lane);
        v16bf b2 = load_fragB(pb[2] + k, lane);
        v16bf b3 = load_fragB(pb[3] + k, lane);
        acc[0] = __builtin_amdgcn_wmma_f32_16x16x32_bf16(false, a, false, b0, (short)0, acc[0], false, false);
        acc[1] = __builtin_amdgcn_wmma_f32_16x16x32_bf16(false, a, false, b1, (short)0, acc[1], false, false);
        acc[2] = __builtin_amdgcn_wmma_f32_16x16x32_bf16(false, a, false, b2, (short)0, acc[2], false, false);
        acc[3] = __builtin_amdgcn_wmma_f32_16x16x32_bf16(false, a, false, b3, (short)0, acc[3], false, false);
    }

    const int mbase = tM + ((lane >> 4) << 3);
#pragma unroll
    for (int j = 0; j < 4; ++j) {
        if (bn[j] >= Nrows) continue;              // divergence only post-WMMA
#pragma unroll
        for (int r = 0; r < 8; ++r) {
            const int m = mbase + r;
            float v = acc[j][r];
            if (resid) v += resid[(size_t)m * ldr + bn[j]];
            if (act == 1) v = geluf(v);
            if (Cf) Cf[(size_t)m * ldc + bn[j]] = v;
            if (Cb) Cb[(size_t)m * ldc + bn[j]] = f2bf(v);
        }
    }
}

// ---------------- f32 -> bf16 convert with K-padding ----------------
__global__ void cvt_pad_kernel(const float* __restrict__ src, __bf16* __restrict__ dst,
                               int R, int K, int Kp) {
    long i = (long)blockIdx.x * blockDim.x + threadIdx.x;
    long n = (long)R * Kp;
    if (i >= n) return;
    int r = (int)(i / Kp), c = (int)(i - (long)r * Kp);
    dst[i] = f2bf(c < K ? src[(size_t)r * K + c] : 0.0f);
}

// ---------------- embeddings ----------------
__global__ void embed_kernel(const int* __restrict__ idx,
                             const float* __restrict__ tok_emb,
                             const float* __restrict__ pos_emb,
                             float* __restrict__ x) {
    int i = blockIdx.x * blockDim.x + threadIdx.x;
    if (i >= MM * DD) return;
    int tok = i / DD, d = i - tok * DD, t = tok % TT;
    x[i] = tok_emb[(size_t)idx[tok] * DD + d] + pos_emb[(size_t)t * DD + d];
}

// ---------------- layernorm (token per thread), outputs f32 + padded bf16 ----------------
__global__ void ln_kernel(const float* __restrict__ x, const float* __restrict__ g,
                          const float* __restrict__ b,
                          float* __restrict__ outf, __bf16* __restrict__ outb) {
    int tok = blockIdx.x * blockDim.x + threadIdx.x;
    if (tok >= MM) return;
    const float* r = x + (size_t)tok * DD;
    float mu = 0.f;
    for (int d = 0; d < DD; ++d) mu += r[d];
    mu *= (1.0f / DD);
    float var = 0.f;
    for (int d = 0; d < DD; ++d) { float u = r[d] - mu; var += u * u; }
    var *= (1.0f / DD);
    float is = rsqrtf(var + 1e-5f);
    for (int d = 0; d < DD; ++d) {
        float z = (r[d] - mu) * is * g[d] + b[d];
        if (outf) outf[(size_t)tok * DD + d] = z;
        outb[(size_t)tok * DP + d] = f2bf(z);
    }
    for (int d = DD; d < DP; ++d) outb[(size_t)tok * DP + d] = f2bf(0.f);
}

// ---------------- causal attention, one (b,h) per block ----------------
__global__ void attn_kernel(const float* __restrict__ q, const float* __restrict__ k,
                            const float* __restrict__ v, __bf16* __restrict__ ob) {
    const int bh = blockIdx.x, b = bh / NHh, h = bh % NHh;
    const int t = threadIdx.x;
    if (h == 0 && t < TT)
        for (int c = DD; c < DP; ++c) ob[(size_t)(b * TT + t) * DP + c] = f2bf(0.f);
    if (t >= TT) return;

    const float scale = 0.0833333333f;             // 144^-0.5
    float qr[HDh];
    const float* qp = q + (size_t)(b * TT + t) * DD + h * HDh;
#pragma unroll
    for (int d = 0; d < HDh; ++d) qr[d] = qp[d];

    float mx = -1e30f;
    for (int s = 0; s <= t; ++s) {
        const float* kp = k + (size_t)(b * TT + s) * DD + h * HDh;
        float acc = 0.f;
#pragma unroll
        for (int d = 0; d < HDh; ++d) acc += qr[d] * kp[d];
        acc *= scale;
        mx = fmaxf(mx, acc);
    }
    float se = 0.f;
    for (int s = 0; s <= t; ++s) {
        const float* kp = k + (size_t)(b * TT + s) * DD + h * HDh;
        float acc = 0.f;
#pragma unroll
        for (int d = 0; d < HDh; ++d) acc += qr[d] * kp[d];
        se += expf(acc * scale - mx);
    }
    float o[HDh];
#pragma unroll
    for (int d = 0; d < HDh; ++d) o[d] = 0.f;
    for (int s = 0; s <= t; ++s) {
        const float* kp = k + (size_t)(b * TT + s) * DD + h * HDh;
        const float* vp = v + (size_t)(b * TT + s) * DD + h * HDh;
        float acc = 0.f;
#pragma unroll
        for (int d = 0; d < HDh; ++d) acc += qr[d] * kp[d];
        float w = expf(acc * scale - mx);
#pragma unroll
        for (int d = 0; d < HDh; ++d) o[d] += w * vp[d];
    }
    float inv = 1.0f / se;
    __bf16* op = ob + (size_t)(b * TT + t) * DP + h * HDh;
#pragma unroll
    for (int d = 0; d < HDh; ++d) op[d] = f2bf(o[d] * inv);
}

// ---------------- noisy top-2 router ----------------
__global__ void router_kernel(const float* __restrict__ h2,
                              const float* __restrict__ rw, const float* __restrict__ rb,
                              const float* __restrict__ nw, const float* __restrict__ nb,
                              float* __restrict__ gate, float* __restrict__ counts,
                              int layer) {
    int tok = blockIdx.x * blockDim.x + threadIdx.x;
    if (tok >= MM) return;
    const float* hr = h2 + (size_t)tok * DD;
    float noisy[NEe];
#pragma unroll
    for (int e = 0; e < NEe; ++e) {
        float lg = rb[e], nl = nb[e];
        const float* rwe = rw + e * DD;
        const float* nwe = nw + e * DD;
        for (int d = 0; d < DD; ++d) { lg += hr[d] * rwe[d]; nl += hr[d] * nwe[d]; }
        float sp = (nl > 20.f) ? nl : log1pf(expf(nl));
        uint32_t seed = ((uint32_t)(layer * MM + tok) * 8u + (uint32_t)e) * 2654435761u + 12345u;
        noisy[e] = lg + hashnorm(seed) * sp;
    }
    int i1 = 0; float v1 = noisy[0];
#pragma unroll
    for (int e = 1; e < NEe; ++e) if (noisy[e] > v1) { v1 = noisy[e]; i1 = e; }
    int i2 = -1; float v2 = -1e30f;
#pragma unroll
    for (int e = 0; e < NEe; ++e) if (e != i1 && noisy[e] > v2) { v2 = noisy[e]; i2 = e; }
    float g1 = 1.0f / (1.0f + expf(v2 - v1));
    float g2 = 1.0f - g1;
#pragma unroll
    for (int e = 0; e < NEe; ++e) gate[(size_t)tok * NEe + e] = 0.f;
    gate[(size_t)tok * NEe + i1] = g1;
    gate[(size_t)tok * NEe + i2] = g2;
    atomicAdd(&counts[i1], 1.f);
    atomicAdd(&counts[i2], 1.f);
}

__global__ void zero_counts_kernel(float* counts) {
    if (threadIdx.x < NEe) counts[threadIdx.x] = 0.f;
}
__global__ void zero_scalar_kernel(float* p) { if (threadIdx.x == 0) *p = 0.f; }

__global__ void aux_kernel(const float* __restrict__ counts, float* __restrict__ auxout) {
    if (threadIdx.x != 0) return;
    float tot = 0.f;
    for (int e = 0; e < NEe; ++e) tot += counts[e];
    const float uni = 1.0f / NEe;
    float aux = 0.f;
    for (int e = 0; e < NEe; ++e) {
        float p = (counts[e] + 1e-3f) / (tot + 1e-3f * NEe);
        aux += uni * (logf(uni) - logf(p + 1e-10f));
    }
    aux /= (float)NEe;
    atomicAdd(auxout, 0.02f * aux);
}

// ---------------- per-expert LN + gated combine ----------------
__global__ void moe_combine_kernel(const float* __restrict__ h2, const float* __restrict__ y,
                                   const float* __restrict__ gate,
                                   const float* __restrict__ eg, const float* __restrict__ eb,
                                   float* __restrict__ x, int e) {
    int tok = blockIdx.x * blockDim.x + threadIdx.x;
    if (tok >= MM) return;
    float g = gate[(size_t)tok * NEe + e];
    if (g == 0.f) return;
    const float* a = h2 + (size_t)tok * DD;
    const float* c = y  + (size_t)tok * DD;
    float mu = 0.f;
    for (int d = 0; d < DD; ++d) mu += a[d] + c[d];
    mu *= (1.0f / DD);
    float var = 0.f;
    for (int d = 0; d < DD; ++d) { float u = a[d] + c[d] - mu; var += u * u; }
    var *= (1.0f / DD);
    float is = rsqrtf(var + 1e-5f);
    const float* ge = eg + (size_t)e * DD;
    const float* be = eb + (size_t)e * DD;
    for (int d = 0; d < DD; ++d) {
        float z = (a[d] + c[d] - mu) * is * ge[d] + be[d];
        x[(size_t)tok * DD + d] += g * z;
    }
}

// ---------------- host side ----------------
static inline size_t alignup(size_t v) { return (v + 255) & ~(size_t)255; }

extern "C" void kernel_launch(void* const* d_in, const int* in_sizes, int n_in,
                              void* d_out, int out_size, void* d_ws, size_t ws_size,
                              hipStream_t stream) {
    (void)in_sizes; (void)n_in; (void)out_size; (void)ws_size;
    // jax pytree flatten order: idx, then params dict sorted:
    // blocks[0..5] (19 sorted keys each), lm_b, lm_w, lnf_b, lnf_g, pos_emb, tok_emb
    const int* idx = (const int*)d_in[0];
    auto BP = [&](int l, int j) -> const float* { return (const float*)d_in[1 + 19 * l + j]; };
    // per-block sorted keys:
    // 0 b1, 1 b2, 2 eln_b, 3 eln_g, 4 ln1_b, 5 ln1_g, 6 ln2_b, 7 ln2_g,
    // 8 noise_b, 9 noise_w, 10 proj_b, 11 proj_w, 12 route_b, 13 route_w,
    // 14 w1, 15 w2, 16 wk, 17 wq, 18 wv
    const float* lm_b    = (const float*)d_in[115];
    const float* lm_w    = (const float*)d_in[116];
    const float* lnf_b   = (const float*)d_in[117];
    const float* lnf_g   = (const float*)d_in[118];
    const float* pos_emb = (const float*)d_in[119];
    const float* tok_emb = (const float*)d_in[120];
    float* dout = (float*)d_out;

    // workspace carve
    char* base = (char*)d_ws;
    size_t off = 0;
    auto carve = [&](size_t bytes) -> char* {
        char* p = base + off; off += alignup(bytes); return p;
    };
    float*  x    = (float*) carve((size_t)MM * DD * 4);
    __bf16* hb   = (__bf16*)carve((size_t)MM * DP * 2);
    float*  qf   = (float*) carve((size_t)MM * DD * 4);
    float*  kf   = (float*) carve((size_t)MM * DD * 4);
    float*  vf   = (float*) carve((size_t)MM * DD * 4);
    __bf16* ob   = (__bf16*)carve((size_t)MM * DP * 2);
    float*  h2   = (float*) carve((size_t)MM * DD * 4);
    __bf16* h2b  = (__bf16*)carve((size_t)MM * DP * 2);
    float*  gate = (float*) carve((size_t)MM * NEe * 4);
    float*  cnts = (float*) carve(256);
    __bf16* hidb = (__bf16*)carve((size_t)MM * H4 * 2);
    float*  yf   = (float*) carve((size_t)MM * DD * 4);
    __bf16* wqb  = (__bf16*)carve((size_t)NLl * DD * DP * 2);
    __bf16* wkb  = (__bf16*)carve((size_t)NLl * DD * DP * 2);
    __bf16* wvb  = (__bf16*)carve((size_t)NLl * DD * DP * 2);
    __bf16* wpb  = (__bf16*)carve((size_t)NLl * DD * DP * 2);
    __bf16* w1b  = (__bf16*)carve((size_t)NLl * NEe * H4 * DP * 2);
    __bf16* w2b  = (__bf16*)carve((size_t)NLl * NEe * DD * H4 * 2);
    __bf16* lmwb = (__bf16*)carve((size_t)VV * DP * 2);

    auto cvt = [&](const float* src, __bf16* dst, int R, int K, int Kp) {
        long n = (long)R * Kp;
        hipLaunchKernelGGL(cvt_pad_kernel, dim3((unsigned)((n + 255) / 256)), dim3(256),
                           0, stream, src, dst, R, K, Kp);
    };
    auto gemm = [&](const __bf16* A, int lda, const __bf16* W, int ldw,
                    const float* bias, const float* resid, int ldr,
                    float* Cf, __bf16* Cb, int ldc, int Mr, int Nr, int K, int act) {
        int stripsN = (Nr + 63) / 64;
        long waves = (long)((Mr + 15) / 16) * stripsN;
        dim3 grid((unsigned)((waves + 7) / 8)), blk(256);
        if (K == DP) {
            hipLaunchKernelGGL(HIP_KERNEL_NAME(gemm_wmma_kernel<DP / 32>), grid, blk, 0, stream,
                               A, lda, W, ldw, bias, resid, ldr, Cf, Cb, ldc, Mr, Nr, stripsN, act);
        } else {   // K == H4
            hipLaunchKernelGGL(HIP_KERNEL_NAME(gemm_wmma_kernel<H4 / 32>), grid, blk, 0, stream,
                               A, lda, W, ldw, bias, resid, ldr, Cf, Cb, ldc, Mr, Nr, stripsN, act);
        }
    };

    // convert all weights to bf16 (done every call; no cached state)
    for (int l = 0; l < NLl; ++l) {
        cvt(BP(l, 17), wqb + (size_t)l * DD * DP, DD, DD, DP);            // wq
        cvt(BP(l, 16), wkb + (size_t)l * DD * DP, DD, DD, DP);            // wk
        cvt(BP(l, 18), wvb + (size_t)l * DD * DP, DD, DD, DP);            // wv
        cvt(BP(l, 11), wpb + (size_t)l * DD * DP, DD, DD, DP);            // proj_w
        cvt(BP(l, 14), w1b + (size_t)l * NEe * H4 * DP, NEe * H4, DD, DP);// w1
        cvt(BP(l, 15), w2b + (size_t)l * NEe * DD * H4, NEe * DD, H4, H4);// w2
    }
    cvt(lm_w, lmwb, VV, DD, DP);

    // aux accumulator lives in d_out[M*V]
    float* auxout = dout + (size_t)MM * VV;
    hipLaunchKernelGGL(zero_scalar_kernel, dim3(1), dim3(32), 0, stream, auxout);

    // embeddings
    hipLaunchKernelGGL(embed_kernel, dim3((MM * DD + 255) / 256), dim3(256), 0, stream,
                       idx, tok_emb, pos_emb, x);

    const dim3 tokGrid((MM + 255) / 256), tokBlk(256);
    for (int l = 0; l < NLl; ++l) {
        // --- attention ---
        hipLaunchKernelGGL(ln_kernel, tokGrid, tokBlk, 0, stream,
                           x, BP(l, 5), BP(l, 4), (float*)nullptr, hb);
        gemm(hb, DP, wqb + (size_t)l * DD * DP, DP, nullptr, nullptr, 0,
             qf, nullptr, DD, MM, DD, DP, 0);
        gemm(hb, DP, wkb + (size_t)l * DD * DP, DP, nullptr, nullptr, 0,
             kf, nullptr, DD, MM, DD, DP, 0);
        gemm(hb, DP, wvb + (size_t)l * DD * DP, DP, nullptr, nullptr, 0,
             vf, nullptr, DD, MM, DD, DP, 0);
        hipLaunchKernelGGL(attn_kernel, dim3(BB * NHh), dim3(64), 0, stream, qf, kf, vf, ob);
        gemm(ob, DP, wpb + (size_t)l * DD * DP, DP, BP(l, 10), x, DD,
             x, nullptr, DD, MM, DD, DP, 0);
        // --- MoE ---
        hipLaunchKernelGGL(ln_kernel, tokGrid, tokBlk, 0, stream,
                           x, BP(l, 7), BP(l, 6), h2, h2b);
        hipLaunchKernelGGL(zero_counts_kernel, dim3(1), dim3(32), 0, stream, cnts);
        hipLaunchKernelGGL(router_kernel, tokGrid, tokBlk, 0, stream,
                           h2, BP(l, 13), BP(l, 12), BP(l, 9), BP(l, 8), gate, cnts, l);
        hipLaunchKernelGGL(aux_kernel, dim3(1), dim3(32), 0, stream, cnts, auxout);
        for (int e = 0; e < NEe; ++e) {
            const __bf16* w1e = w1b + ((size_t)l * NEe + e) * H4 * DP;
            const __bf16* w2e = w2b + ((size_t)l * NEe + e) * DD * H4;
            gemm(h2b, DP, w1e, DP, BP(l, 0) + (size_t)e * H4, nullptr, 0,
                 nullptr, hidb, H4, MM, H4, DP, 1);                     // gelu, bf16 out
            gemm(hidb, H4, w2e, H4, BP(l, 1) + (size_t)e * DD, nullptr, 0,
                 yf, nullptr, DD, MM, DD, H4, 0);
            hipLaunchKernelGGL(moe_combine_kernel, tokGrid, tokBlk, 0, stream,
                               h2, yf, gate, BP(l, 3), BP(l, 2), x, e);
        }
    }

    // final LN + LM head -> logits in d_out
    hipLaunchKernelGGL(ln_kernel, tokGrid, tokBlk, 0, stream,
                       x, lnf_g, lnf_b, (float*)nullptr, hb);
    gemm(hb, DP, lmwb, DP, lm_b, nullptr, 0, dout, nullptr, VV, MM, VV, DP, 0);
}
